// tensoRF_35751307772024
// MI455X (gfx1250) — matrix operand order repaired and checked by reading, not compiled, not run on hardware
//
#include <hip/hip_runtime.h>
#include <hip/hip_bf16.h>

typedef __attribute__((ext_vector_type(16))) _Float16 v16h;
typedef __attribute__((ext_vector_type(8)))  float    v8f;

#define RS 300
#define NPTS (1 << 20)

__device__ __forceinline__ int iclamp(int v, int lo, int hi) {
    return v < lo ? lo : (v > hi ? hi : v);
}

__global__ __launch_bounds__(256) void tensorf_fused_kernel(
    const float* __restrict__ pts,      // (N,3)
    const float* __restrict__ dplane,   // (3,8,RS,RS)
    const float* __restrict__ dline,    // (3,8,RS)
    const float* __restrict__ aplane,   // (3,16,RS,RS)
    const float* __restrict__ aline,    // (3,16,RS)
    const float* __restrict__ bw,       // (3,48)
    float* __restrict__ out)            // sigma[N] ++ app[N*3]
{
    // Feature staging for WMMA: 256 points x 64 halves (48 real + 16 zero pad),
    // row stride 72 halves to spread LDS banks.
    __shared__ _Float16 feat_lds[256][72];

    const int tid = threadIdx.x;
    const int n   = blockIdx.x * 256 + tid;

    // ---- normalized grid coords ----
    const float p0 = pts[n * 3 + 0];
    const float p1 = pts[n * 3 + 1];
    const float p2 = pts[n * 3 + 2];
    const float g0 = (p0 + 0.125f) * 8.0f - 1.0f;
    const float g1 = (p1 - 0.750f) * 8.0f - 1.0f;
    const float g2 = (p2 + 0.125f) * 8.0f - 1.0f;

    const float gx[3] = { g0, g0, g1 };
    const float gy[3] = { g1, g2, g2 };
    const float gl[3] = { g2, g1, g0 };

    float sigma = 0.0f;
    float feat[48];

    #pragma unroll
    for (int p = 0; p < 3; ++p) {
        // ---- plane bilinear setup (shared by all 24 channels of plane p) ----
        const float fx = (gx[p] + 1.0f) * 0.5f * (float)(RS - 1);
        const float fy = (gy[p] + 1.0f) * 0.5f * (float)(RS - 1);
        const int x0 = iclamp((int)floorf(fx), 0, RS - 1);
        const int y0 = iclamp((int)floorf(fy), 0, RS - 1);
        const int x1 = x0 + 1 > RS - 1 ? RS - 1 : x0 + 1;
        const int y1 = y0 + 1 > RS - 1 ? RS - 1 : y0 + 1;
        const float wx = fx - (float)x0;
        const float wy = fy - (float)y0;
        const float w00 = (1.0f - wx) * (1.0f - wy);
        const float w01 = wx * (1.0f - wy);
        const float w10 = (1.0f - wx) * wy;
        const float w11 = wx * wy;
        const int o00 = y0 * RS + x0;
        const int o01 = y0 * RS + x1;
        const int o10 = y1 * RS + x0;
        const int o11 = y1 * RS + x1;

        // ---- line lerp setup ----
        const float fl = (gl[p] + 1.0f) * 0.5f * (float)(RS - 1);
        const int l0 = iclamp((int)floorf(fl), 0, RS - 1);
        const int l1 = l0 + 1 > RS - 1 ? RS - 1 : l0 + 1;
        const float wl = fl - (float)l0;

        // ---- density: 8 channels -> sigma ----
        {
            const float* __restrict__ pb = dplane + (size_t)p * 8 * RS * RS;
            const float* __restrict__ lb = dline  + (size_t)p * 8 * RS;
            #pragma unroll
            for (int c = 0; c < 8; ++c) {
                const float* __restrict__ pc = pb + (size_t)c * RS * RS;
                const float v = w00 * pc[o00] + w01 * pc[o01]
                              + w10 * pc[o10] + w11 * pc[o11];
                const float* __restrict__ lc = lb + c * RS;
                const float lv = lc[l0] * (1.0f - wl) + lc[l1] * wl;
                sigma += v * lv;
            }
        }

        // ---- appearance: 16 channels -> feat ----
        {
            const float* __restrict__ pb = aplane + (size_t)p * 16 * RS * RS;
            const float* __restrict__ lb = aline  + (size_t)p * 16 * RS;
            #pragma unroll
            for (int c = 0; c < 16; ++c) {
                const float* __restrict__ pc = pb + (size_t)c * RS * RS;
                const float v = w00 * pc[o00] + w01 * pc[o01]
                              + w10 * pc[o10] + w11 * pc[o11];
                const float* __restrict__ lc = lb + c * RS;
                const float lv = lc[l0] * (1.0f - wl) + lc[l1] * wl;
                feat[p * 16 + c] = v * lv;
            }
        }
    }

    out[n] = sigma;

    // ---- stage features to LDS (f16), zero-pad K to 64 ----
    #pragma unroll
    for (int k = 0; k < 48; ++k) feat_lds[tid][k] = (_Float16)feat[k];
    #pragma unroll
    for (int k = 48; k < 64; ++k) feat_lds[tid][k] = (_Float16)0.0f;
    __syncthreads();

    // ---- WMMA projection: app(16x16 per tile) = basis(16x64 f16) x feat(64x16 f16) ----
    const int lane = tid & 31;
    const int wave = tid >> 5;
    const int m    = lane & 15;              // A row / B column / C column index
    const int kA   = (lane < 16) ? 0 : 16;   // A-matrix K-half offset
    const int kB   = (lane < 16) ? 0 : 16;   // B-matrix K-half offset

    // Build A (basis weights), identical for both tiles of this wave.
    v16h a0, a1;
    #pragma unroll
    for (int j = 0; j < 8; ++j) {
        const int kk = kA + 2 * j;
        // chunk 0: K = kk, kk+1
        a0[2 * j]     = (_Float16)((m < 3 && kk     < 48) ? bw[m * 48 + kk]     : 0.0f);
        a0[2 * j + 1] = (_Float16)((m < 3 && kk + 1 < 48) ? bw[m * 48 + kk + 1] : 0.0f);
        // chunk 1: K = 32+kk, 33+kk
        const int kk1 = 32 + kk;
        a1[2 * j]     = (_Float16)((m < 3 && kk1     < 48) ? bw[m * 48 + kk1]     : 0.0f);
        a1[2 * j + 1] = (_Float16)((m < 3 && kk1 + 1 < 48) ? bw[m * 48 + kk1 + 1] : 0.0f);
    }

    #pragma unroll
    for (int t = 0; t < 2; ++t) {
        const int tile = wave * 2 + t;       // 16 points per tile, 16 tiles per block
        const int col  = tile * 16 + m;      // point (column) this lane feeds

        v16h b0, b1;
        #pragma unroll
        for (int j = 0; j < 8; ++j) {
            b0[2 * j]     = feat_lds[col][kB + 2 * j];
            b0[2 * j + 1] = feat_lds[col][kB + 2 * j + 1];
            b1[2 * j]     = feat_lds[col][32 + kB + 2 * j];
            b1[2 * j + 1] = feat_lds[col][32 + kB + 2 * j + 1];
        }

        v8f c = {};
        c = __builtin_amdgcn_wmma_f32_16x16x32_f16(false, a0, false, b0,
                                                   (short)0, c, false, false);
        c = __builtin_amdgcn_wmma_f32_16x16x32_f16(false, a1, false, b1,
                                                   (short)0, c, false, false);

        // C layout: VGPR r, lanes 0-15 hold M=r, N=lane. Rows 0..2 are the app outputs.
        if (lane < 16) {
            const int npt = blockIdx.x * 256 + tile * 16 + lane;
            float* __restrict__ app = out + NPTS;
            app[npt * 3 + 0] = c[0];
            app[npt * 3 + 1] = c[1];
            app[npt * 3 + 2] = c[2];
        }
    }
}

extern "C" void kernel_launch(void* const* d_in, const int* in_sizes, int n_in,
                              void* d_out, int out_size, void* d_ws, size_t ws_size,
                              hipStream_t stream) {
    const float* pts    = (const float*)d_in[0];
    const float* dplane = (const float*)d_in[1];
    const float* dline  = (const float*)d_in[2];
    const float* aplane = (const float*)d_in[3];
    const float* aline  = (const float*)d_in[4];
    const float* bw     = (const float*)d_in[5];
    float* out = (float*)d_out;

    const int n_points = in_sizes[0] / 3;          // 1<<20
    const int grid     = (n_points + 255) / 256;   // exact: 4096 blocks

    tensorf_fused_kernel<<<grid, 256, 0, stream>>>(pts, dplane, dline,
                                                   aplane, aline, bw, out);
}
